// MaskedSelfAttention_34059090657395
// MI455X (gfx1250) — matrix operand order repaired and checked by reading, not compiled, not run on hardware
//
#include <hip/hip_runtime.h>

// ---------------------------------------------------------------------------
// Types for CDNA5 (gfx1250) WMMA: 16x16x32 bf16 -> f32
// ---------------------------------------------------------------------------
typedef __bf16 bf16_t;
typedef __attribute__((ext_vector_type(16))) __bf16 v16bf;
typedef __attribute__((ext_vector_type(8)))  __bf16 v8bf;
typedef __attribute__((ext_vector_type(8)))  float   v8f;
typedef __attribute__((__vector_size__(16))) int     vi4_t;  // int4 for async-LDS builtin

union Frag16 { v16bf v; v8bf h[2]; };

__device__ __forceinline__ v8f wmma_bf16(v16bf a, v16bf b, v8f c) {
  // 8 args: (neg_a, A, neg_b, B, c_mod, C, reuse_a, reuse_b)
  return __builtin_amdgcn_wmma_f32_16x16x32_bf16(false, a, false, b, (short)0, c,
                                                 false, false);
}

__device__ __forceinline__ v8f vzero8() {
  v8f z = {0.f, 0.f, 0.f, 0.f, 0.f, 0.f, 0.f, 0.f};
  return z;
}

// Async global->LDS path (CDNA5 GLOBAL_LOAD_ASYNC_TO_LDS_*, tracked by ASYNCcnt)
#if defined(__has_builtin)
#if __has_builtin(__builtin_amdgcn_global_load_async_to_lds_b128)
#define HAVE_ASYNC_LDS 1
#endif
#endif
#ifndef HAVE_ASYNC_LDS
#define HAVE_ASYNC_LDS 0
#endif

__device__ __forceinline__ void async_wait0() {
#if HAVE_ASYNC_LDS
#if __has_builtin(__builtin_amdgcn_s_wait_asynccnt)
  __builtin_amdgcn_s_wait_asynccnt(0);
#else
  asm volatile("s_wait_asynccnt 0x0" ::: "memory");
#endif
#endif
}

// Problem constants
static constexpr int T_DIM = 2048;
static constexpr int C_DIM = 4096;
static constexpr int HS    = 128;
static constexpr int NH    = 32;
static constexpr int NKV   = 8;
static constexpr int QKV_O = (NH + 2 * NKV) * HS;   // 6144
static constexpr float SCALE = 0.08838834764831845f; // 1/sqrt(128)
static constexpr float NEG_BIG = -3.0e38f;

// LDS tile row stride (elements): 32 data + 8 pad -> 80B rows, 16B aligned,
// bank-conflict-free 16B fragment reads.
static constexpr int LDSTR = 40;

// ---------------------------------------------------------------------------
// fp32 -> bf16 conversion (grid-stride)
// ---------------------------------------------------------------------------
__global__ void k_f32_to_bf16(const float* __restrict__ in, bf16_t* __restrict__ out,
                              int n) {
  for (int i = blockIdx.x * blockDim.x + threadIdx.x; i < n;
       i += gridDim.x * blockDim.x) {
    out[i] = (bf16_t)in[i];
  }
}

// ---------------------------------------------------------------------------
// Stage a 128x32 bf16 tile (row stride K in global) into LDS (row stride 40).
// 512 x 16B chunks, 2 per thread; async DMA when available.
// ---------------------------------------------------------------------------
__device__ __forceinline__ void stage_tile(const bf16_t* __restrict__ g,
                                           bf16_t* l, int K, int tid) {
#pragma unroll
  for (int c = 0; c < 2; ++c) {
    const int chunk = tid + c * 256;          // 0..511
    const int row = chunk >> 2;               // 0..127
    const int seg = chunk & 3;                // 16B segment within 64B row
    const bf16_t* gp = g + (size_t)row * K + seg * 8;
    bf16_t* lp = l + row * LDSTR + seg * 8;
#if HAVE_ASYNC_LDS
    __builtin_amdgcn_global_load_async_to_lds_b128(
        (__attribute__((address_space(1))) vi4_t*)gp,
        (__attribute__((address_space(3))) vi4_t*)lp, 0, 0);
#else
    *(v8bf*)lp = *(const v8bf*)gp;
#endif
  }
}

// ---------------------------------------------------------------------------
// Generic GEMM: C[M,N] (f32) = A[M,K] (bf16, row-major) * B[N,K]^T (bf16, row-major)
// Block: 256 threads = 8 waves, 128x128 tile, K-step 32, double-buffered LDS
// staging fed by async global->LDS DMA. Wave grid 2(M) x 4(N); each wave does
// 4x2 WMMA accumulators (64x32). Fragments read from LDS in the documented
// 16-bit A/B VGPR layouts.
// ---------------------------------------------------------------------------
__global__ __launch_bounds__(256)
void k_gemm_bf16_f32(const bf16_t* __restrict__ A, const bf16_t* __restrict__ B,
                     float* __restrict__ C, int M, int N, int K) {
  __shared__ __align__(16) bf16_t lA[2][128 * LDSTR];
  __shared__ __align__(16) bf16_t lB[2][128 * LDSTR];

  const int tid   = threadIdx.x;
  const int wid   = tid >> 5;
  const int lane  = tid & 31;
  const int half  = lane >> 4;       // 0/1
  const int r     = lane & 15;       // row (A) / column (B) within 16
  const int waveM = wid >> 2;        // 0..1
  const int waveN = wid & 3;         // 0..3
  const int bm    = blockIdx.y * 128;
  const int bn    = blockIdx.x * 128;

  const bf16_t* Ab = A + (size_t)bm * K;
  const bf16_t* Bb = B + (size_t)bn * K;

  v8f acc[4][2];
#pragma unroll
  for (int mi = 0; mi < 4; ++mi)
#pragma unroll
    for (int ni = 0; ni < 2; ++ni) acc[mi][ni] = vzero8();

  const int nK = K >> 5;

  stage_tile(Ab, lA[0], K, tid);
  stage_tile(Bb, lB[0], K, tid);
  async_wait0();
  __syncthreads();

  for (int kt = 0; kt < nK; ++kt) {
    const int cur = kt & 1;
    if (kt + 1 < nK) {
      stage_tile(Ab + (kt + 1) * 32, lA[cur ^ 1], K, tid);
      stage_tile(Bb + (kt + 1) * 32, lB[cur ^ 1], K, tid);
    }

    Frag16 af[4], bfv[2];
    const bf16_t* la = lA[cur];
    const bf16_t* lb = lB[cur];
#pragma unroll
    for (int mi = 0; mi < 4; ++mi) {
      const bf16_t* p = la + (waveM * 64 + mi * 16 + r) * LDSTR + 8 * half;
      af[mi].h[0] = *(const v8bf*)p;          // K = 8h .. 8h+7
      af[mi].h[1] = *(const v8bf*)(p + 16);   // K = 16+8h .. 16+8h+7
    }
#pragma unroll
    for (int ni = 0; ni < 2; ++ni) {
      const bf16_t* p = lb + (waveN * 32 + ni * 16 + r) * LDSTR + 8 * half;
      bfv[ni].h[0] = *(const v8bf*)p;         // K = 8h .. 8h+7
      bfv[ni].h[1] = *(const v8bf*)(p + 16);  // K = 16+8h .. 16+8h+7
    }
#pragma unroll
    for (int mi = 0; mi < 4; ++mi)
#pragma unroll
      for (int ni = 0; ni < 2; ++ni)
        acc[mi][ni] = wmma_bf16(af[mi].v, bfv[ni].v, acc[mi][ni]);

    if (kt + 1 < nK) {
      async_wait0();          // next tile fully landed in LDS
      __syncthreads();        // and everyone is done reading the current one
    }
  }

#pragma unroll
  for (int mi = 0; mi < 4; ++mi)
#pragma unroll
    for (int ni = 0; ni < 2; ++ni)
#pragma unroll
      for (int vv = 0; vv < 8; ++vv) {
        const int row = bm + waveM * 64 + mi * 16 + vv + 8 * half;
        const int col = bn + waveN * 32 + ni * 16 + r;
        C[(size_t)row * N + col] = acc[mi][ni][vv];
      }
}

// ---------------------------------------------------------------------------
// RoPE + scatter: qkv f32 [T][6144] -> qb bf16 [NH][T][128] (roped),
// kb bf16 [NKV][T][128] (roped), vT bf16 [NKV][128][T] (transposed for P*V).
// ---------------------------------------------------------------------------
__global__ void k_rope_scatter(const float* __restrict__ qkv,
                               const float* __restrict__ cosb,
                               const float* __restrict__ sinb,
                               bf16_t* __restrict__ qb, bf16_t* __restrict__ kb,
                               bf16_t* __restrict__ vT) {
  const int total = T_DIM * QKV_O;
  for (int idx = blockIdx.x * blockDim.x + threadIdx.x; idx < total;
       idx += gridDim.x * blockDim.x) {
    const int t = idx / QKV_O;
    const int o = idx - t * QKV_O;
    const int g    = o / 768;          // kv group
    const int rem  = o - g * 768;
    const int slot = rem >> 7;         // 0..5
    const int d    = rem & 127;
    const float val = qkv[idx];
    if (slot == 5) {
      vT[((size_t)(g * HS + d)) * T_DIM + t] = (bf16_t)val;
      continue;
    }
    const int base = t * QKV_O + g * 768 + slot * 128;
    float outv;
    if (d < 64) {
      const float c = cosb[t * 64 + d], s = sinb[t * 64 + d];
      outv = val * c - qkv[base + 64 + d] * s;
    } else {
      const int dd = d - 64;
      const float c = cosb[t * 64 + dd], s = sinb[t * 64 + dd];
      outv = val * c + qkv[base + dd] * s;
    }
    if (slot == 4)
      kb[((size_t)g * T_DIM + t) * HS + d] = (bf16_t)outv;
    else
      qb[((size_t)(g * 4 + slot) * T_DIM + t) * HS + d] = (bf16_t)outv;
  }
}

// ---------------------------------------------------------------------------
// Flash attention: one block per (head, 128-row query tile). 8 waves; each
// wave owns 16 query rows, streams 32 keys/chunk. Scores via 8 WMMAs (K=128),
// online softmax (shfl_xor reductions in 16-lane groups), P re-laid-out
// through per-wave LDS into A-fragment form, then 8 P*V WMMAs per chunk.
// Next key chunk is prefetched (global_prefetch_b8).
// ---------------------------------------------------------------------------
__global__ __launch_bounds__(256)
void k_attn(const bf16_t* __restrict__ qb, const bf16_t* __restrict__ kb,
            const bf16_t* __restrict__ vT, bf16_t* __restrict__ yb) {
  __shared__ __align__(16) bf16_t plds[8][16 * 32];  // 1 KB per wave

  const int head  = blockIdx.y;        // 0..31
  const int g     = head >> 2;         // kv group
  const int tTile = blockIdx.x * 128;
  const int wid   = threadIdx.x >> 5;
  const int lane  = threadIdx.x & 31;
  const int half  = lane >> 4;
  const int r     = lane & 15;
  const int row0  = tTile + wid * 16;  // this wave's first query row
  const int rowHi = row0 + 15;

  // Q fragments for this wave's 16 rows, all 128 dims (4 chunks of K=32)
  Frag16 qf[4];
#pragma unroll
  for (int kc = 0; kc < 4; ++kc) {
    const bf16_t* p =
        qb + ((size_t)head * T_DIM + row0 + r) * HS + kc * 32 + 8 * half;
    qf[kc].h[0] = *(const v8bf*)p;
    qf[kc].h[1] = *(const v8bf*)(p + 16);
  }

  v8f acc[8];
#pragma unroll
  for (int nt = 0; nt < 8; ++nt) acc[nt] = vzero8();
  float mrow[8], lrow[8];
#pragma unroll
  for (int vv = 0; vv < 8; ++vv) { mrow[vv] = NEG_BIG; lrow[vv] = 0.f; }

  for (int kt = 0; kt <= rowHi; kt += 32) {
    // prefetch next chunk's key rows while we compute on this one
    if (kt + 32 <= rowHi) {
      __builtin_prefetch(kb + ((size_t)g * T_DIM + kt + 32 + r) * HS, 0, 3);
      __builtin_prefetch(kb + ((size_t)g * T_DIM + kt + 48 + r) * HS, 0, 3);
    }

    // ---- scores: two 16x16 tiles (keys kt..kt+15, kt+16..kt+31) ----
    v8f s0 = vzero8(), s1 = vzero8();
#pragma unroll
    for (int kc = 0; kc < 4; ++kc) {
      Frag16 kf;
      const bf16_t* p0 =
          kb + ((size_t)g * T_DIM + kt + r) * HS + kc * 32 + 8 * half;
      kf.h[0] = *(const v8bf*)p0;
      kf.h[1] = *(const v8bf*)(p0 + 16);
      s0 = wmma_bf16(qf[kc].v, kf.v, s0);
      const bf16_t* p1 =
          kb + ((size_t)g * T_DIM + kt + 16 + r) * HS + kc * 32 + 8 * half;
      kf.h[0] = *(const v8bf*)p1;
      kf.h[1] = *(const v8bf*)(p1 + 16);
      s1 = wmma_bf16(qf[kc].v, kf.v, s1);
    }

    // ---- online softmax update ----
    const int key0 = kt + r, key1 = kt + 16 + r;
#pragma unroll
    for (int vv = 0; vv < 8; ++vv) {
      const int row = row0 + vv + 8 * half;
      float a0 = (key0 <= row) ? s0[vv] * SCALE : NEG_BIG;
      float a1 = (key1 <= row) ? s1[vv] * SCALE : NEG_BIG;
      float mx = fmaxf(a0, a1);
#pragma unroll
      for (int off = 8; off > 0; off >>= 1)
        mx = fmaxf(mx, __shfl_xor(mx, off, 32));
      const float mnew  = fmaxf(mrow[vv], mx);
      const float alpha = __expf(mrow[vv] - mnew);
      mrow[vv] = mnew;
      const float p0 = __expf(a0 - mnew);
      const float p1 = __expf(a1 - mnew);
      float sum = p0 + p1;
#pragma unroll
      for (int off = 8; off > 0; off >>= 1)
        sum += __shfl_xor(sum, off, 32);
      lrow[vv] = lrow[vv] * alpha + sum;
#pragma unroll
      for (int nt = 0; nt < 8; ++nt) acc[nt][vv] *= alpha;
      // stage P (bf16) into this wave's LDS tile, row-major [16][32]
      plds[wid][(vv + 8 * half) * 32 + r]      = (bf16_t)p0;
      plds[wid][(vv + 8 * half) * 32 + 16 + r] = (bf16_t)p1;
    }

    // ---- reload P in A-matrix (16x32) layout ----
    Frag16 pf;
    const bf16_t* lp = &plds[wid][0];
    pf.h[0] = *(const v8bf*)(lp + r * 32 + 8 * half);
    pf.h[1] = *(const v8bf*)(lp + r * 32 + 16 + 8 * half);

    // ---- acc += P (16x32) * V (32x128) ----
#pragma unroll
    for (int nt = 0; nt < 8; ++nt) {
      Frag16 vf;
      const bf16_t* pv =
          vT + ((size_t)(g * HS + nt * 16 + r)) * T_DIM + kt + 8 * half;
      vf.h[0] = *(const v8bf*)pv;
      vf.h[1] = *(const v8bf*)(pv + 16);
      acc[nt] = wmma_bf16(pf.v, vf.v, acc[nt]);
    }
  }

  // ---- epilogue: normalize and store y[t][head*128 + d] ----
#pragma unroll
  for (int nt = 0; nt < 8; ++nt) {
    const int col = head * HS + nt * 16 + r;
#pragma unroll
    for (int vv = 0; vv < 8; ++vv) {
      const int row = row0 + vv + 8 * half;
      yb[(size_t)row * C_DIM + col] = (bf16_t)(acc[nt][vv] / lrow[vv]);
    }
  }
}

// ---------------------------------------------------------------------------
// Host-side launch
// ---------------------------------------------------------------------------
extern "C" void kernel_launch(void* const* d_in, const int* in_sizes, int n_in,
                              void* d_out, int out_size, void* d_ws,
                              size_t ws_size, hipStream_t stream) {
  (void)in_sizes; (void)n_in; (void)out_size; (void)ws_size;
  const float* x      = (const float*)d_in[0];
  const float* cosb   = (const float*)d_in[1];
  const float* sinb   = (const float*)d_in[2];
  // d_in[3] = attn_mask (bool) — causal mask computed analytically, unused
  const float* w_attn = (const float*)d_in[4];
  const float* w_proj = (const float*)d_in[5];
  float* out = (float*)d_out;

  char* ws = (char*)d_ws;
  const size_t SZ_XB  = (size_t)T_DIM * C_DIM * 2;      // 16 MB
  const size_t SZ_WAB = (size_t)QKV_O * C_DIM * 2;      // 48 MB
  const size_t SZ_WPB = (size_t)C_DIM * C_DIM * 2;      // 32 MB
  const size_t SZ_QKV = (size_t)T_DIM * QKV_O * 4;      // 48 MB
  const size_t SZ_QB  = (size_t)NH * T_DIM * HS * 2;    // 16 MB
  const size_t SZ_KB  = (size_t)NKV * T_DIM * HS * 2;   //  4 MB
  const size_t SZ_VT  = (size_t)NKV * HS * T_DIM * 2;   //  4 MB

  bf16_t* xb   = (bf16_t*)(ws);
  bf16_t* wab  = (bf16_t*)(ws + SZ_XB);
  bf16_t* wpb  = (bf16_t*)(ws + SZ_XB + SZ_WAB);
  float*  qkvf = (float*)(ws + SZ_XB + SZ_WAB + SZ_WPB);
  bf16_t* qb   = (bf16_t*)(ws + SZ_XB + SZ_WAB + SZ_WPB + SZ_QKV);
  bf16_t* kb   = (bf16_t*)(ws + SZ_XB + SZ_WAB + SZ_WPB + SZ_QKV + SZ_QB);
  bf16_t* vT   = (bf16_t*)(ws + SZ_XB + SZ_WAB + SZ_WPB + SZ_QKV + SZ_QB + SZ_KB);
  bf16_t* yb   = (bf16_t*)(ws + SZ_XB + SZ_WAB + SZ_WPB + SZ_QKV + SZ_QB + SZ_KB + SZ_VT);

  // 1) fp32 -> bf16 conversions
  k_f32_to_bf16<<<4096, 256, 0, stream>>>(x, xb, T_DIM * C_DIM);
  k_f32_to_bf16<<<8192, 256, 0, stream>>>(w_attn, wab, QKV_O * C_DIM);
  k_f32_to_bf16<<<8192, 256, 0, stream>>>(w_proj, wpb, C_DIM * C_DIM);

  // 2) QKV projection: qkv[T,6144] = x[T,4096] * w_attn[6144,4096]^T
  k_gemm_bf16_f32<<<dim3(QKV_O / 128, T_DIM / 128), 256, 0, stream>>>(
      xb, wab, qkvf, T_DIM, QKV_O, C_DIM);

  // 3) RoPE + scatter into q/k/vT
  k_rope_scatter<<<8192, 256, 0, stream>>>(qkvf, cosb, sinb, qb, kb, vT);

  // 4) Flash attention -> yb[T, C] (bf16)
  k_attn<<<dim3(T_DIM / 128, NH), 256, 0, stream>>>(qb, kb, vT, yb);

  // 5) Output projection: out[T,4096] = y[T,4096] * w_proj[4096,4096]^T
  k_gemm_bf16_f32<<<dim3(C_DIM / 128, T_DIM / 128), 256, 0, stream>>>(
      yb, wpb, out, T_DIM, C_DIM, C_DIM);
}
